// VectorQuantizer_64742337020152
// MI455X (gfx1250) — matrix operand order repaired and checked by reading, not compiled, not run on hardware
//
#include <hip/hip_runtime.h>

// ---------------------------------------------------------------- constants
#define BB   32
#define LL   1024
#define CC   256
#define NE   1024
#define NN   (BB * LL)          // 32768 rows
#define BETA 0.25f

// d_out layout (flat float32, tuple order)
#define O_LOSS 0
#define O_ZQ   1ull
#define O_PERP (1ull + 8388608ull)                  // 8388609
#define O_OH   (2ull + 8388608ull)                  // 8388610
#define O_IDX  (2ull + 8388608ull + 33554432ull)    // 41943042

// workspace layout (byte offsets)
#define WS_ZBF   0ull                         // 8388608 * 2  = 16 MiB
#define WS_EBF   16777216ull                  // 262144 * 2
#define WS_ENORM 17301504ull                  // 1024 * 4
#define WS_IDX   17305600ull                  // 32768 * 4
#define WS_CNT   17436672ull                  // 1024 * 4
#define WS_LOSS  17440768ull                  // 4

typedef __attribute__((ext_vector_type(16))) __bf16       v16bf;
typedef __attribute__((ext_vector_type(8)))  float        v8f;
typedef __attribute__((ext_vector_type(4))) unsigned int  u32x4;
typedef __attribute__((ext_vector_type(8)))  int          i32x8;
typedef __attribute__((ext_vector_type(4)))  int          i32x4;

union Frag16 { u32x4 u[2]; v16bf v; };

static __device__ __forceinline__ unsigned short f2bf(float f) {
    unsigned int u = __float_as_uint(f);
    u += 0x7FFFu + ((u >> 16) & 1u);     // round-to-nearest-even
    return (unsigned short)(u >> 16);
}

// ------------------------------------------------ 1a: embedding -> bf16 + ||e||^2
__global__ __launch_bounds__(256)
void vq_prep_embed(const float* __restrict__ emb,
                   unsigned short* __restrict__ ebf,
                   float* __restrict__ enorm) {
    __shared__ float sred[256];
    const int e = blockIdx.x;
    const int c = threadIdx.x;
    float v = emb[e * CC + c];
    ebf[e * CC + c] = f2bf(v);
    sred[c] = v * v;
    __syncthreads();
    for (int s = 128; s > 0; s >>= 1) {
        if (c < s) sred[c] += sred[c + s];
        __syncthreads();
    }
    if (c == 0) enorm[e] = sred[0];
}

// ------------------------------------------------ 1b: z (B,C,L) -> zbf (N,C) bf16
__global__ __launch_bounds__(256)
void vq_transpose_z(const float* __restrict__ z,
                    unsigned short* __restrict__ zbf) {
    __shared__ float tile[32][33];
    const int b  = blockIdx.z;
    const int c0 = blockIdx.y * 32;
    const int l0 = blockIdx.x * 32;
    const int tx = threadIdx.x, ty = threadIdx.y;
#pragma unroll
    for (int j = 0; j < 4; ++j) {
        int c = c0 + ty + j * 8;
        tile[ty + j * 8][tx] = z[(size_t)b * CC * LL + (size_t)c * LL + l0 + tx];
    }
    __syncthreads();
#pragma unroll
    for (int j = 0; j < 4; ++j) {
        int l = l0 + ty + j * 8;
        zbf[((size_t)b * LL + l) * CC + c0 + tx] = f2bf(tile[tx][ty + j * 8]);
    }
}

// ------------------------------------------------ 2: TDM-staged WMMA GEMM + argmin
// block = 256 threads (8 waves), 128 rows / block; grid = 256 blocks.
// B tiles (128 cols x 256 K, bf16 = 64 KB) are DMA'd into LDS by the Tensor
// Data Mover, double-buffered (128 KB of the 320 KB WGP LDS), overlapped with
// WMMA on the previous tile via TENSORcnt.
__global__ __launch_bounds__(256)
void vq_score_argmin(const unsigned short* __restrict__ zbf,
                     const unsigned short* __restrict__ ebf,
                     const float* __restrict__ enorm,
                     int* __restrict__ d_idx) {
    __shared__ unsigned short bsh[2][128 * 256];   // 2 x 64 KB

    const int lane = threadIdx.x & 31;
    const int wave = threadIdx.x >> 5;
    const int hi   = lane >> 4;      // lane half
    const int ln   = lane & 15;
    const int row0 = blockIdx.x * 128 + wave * 16;
    const bool issuer = (threadIdx.x < 32);        // wave 0 drives the TDM

    // --- Tensor DMA: ebf tile [chunk*128 .. +128) x 256 bf16  ->  bsh[buf]
    auto issue_tdm = [&](int chunk, int buf) {
        unsigned long long ga =
            (unsigned long long)(const void*)(ebf + (size_t)chunk * 128 * 256);
        unsigned lds = (unsigned)(unsigned long long)(const void*)&bsh[buf][0];
        u32x4 g0;
        g0[0] = 1u;                                    // count=1, user mode
        g0[1] = lds;                                   // lds_addr
        g0[2] = (unsigned)ga;                          // global_addr[31:0]
        g0[3] = ((unsigned)(ga >> 32) & 0x1FFFFFFu)    // global_addr[56:32]
                | (2u << 30);                          // type = 2 (image)
        i32x8 g1;
        g1[0] = 0x00010000;          // wg_mask=0, data_size=1 (2-byte elems)
        g1[1] = (int)(256u << 16);   // tensor_dim0 = 256 (bits 63:48)
        g1[2] = (int)(128u << 16);   // tensor_dim1 = 128 (bits 95:80)
        g1[3] = (int)(256u << 16);   // tile_dim0   = 256 (bits 127:112)
        g1[4] = 128;                 // tile_dim1   = 128 (bits 143:128)
        g1[5] = 256;                 // tensor_dim0_stride = 256
        g1[6] = 0;
        g1[7] = 0;
        i32x4 gz4 = (i32x4){0, 0, 0, 0};
        i32x8 gz8 = (i32x8){0, 0, 0, 0, 0, 0, 0, 0};
        __builtin_amdgcn_tensor_load_to_lds(g0, g1, gz4, gz4, gz8, 0);
    };

    // Preload all 8 A fragments (16x32 bf16 each) for this wave's 16-row strip.
    // A layout: lane m = ln, element i -> k = (i%8) + 8*hi + 16*(i/8)
    Frag16 afr[8];
    {
        const unsigned short* arow = zbf + (size_t)(row0 + ln) * CC;
#pragma unroll
        for (int ks = 0; ks < 8; ++ks) {
            const unsigned short* p = arow + ks * 32 + 8 * hi;
            afr[ks].u[0] = *(const u32x4*)(p);
            afr[ks].u[1] = *(const u32x4*)(p + 16);
        }
    }

    float bestv[8];
    int   bestc[8];
#pragma unroll
    for (int r = 0; r < 8; ++r) { bestv[r] = 3.4e38f; bestc[r] = 0; }

    if (issuer) issue_tdm(0, 0);                   // prime the pipeline

    for (int chunk = 0; chunk < 8; ++chunk) {
        if (issuer) {
            if (chunk < 7) {
                issue_tdm(chunk + 1, (chunk + 1) & 1);     // prefetch next tile
                __builtin_amdgcn_s_wait_tensorcnt(1);      // current tile landed
            } else {
                __builtin_amdgcn_s_wait_tensorcnt(0);      // drain
            }
        }
        __syncthreads();                                    // tile visible to WG

        const int col0 = chunk * 128;
        const unsigned short* bbase = &bsh[chunk & 1][0];
        v8f acc[8];
#pragma unroll
        for (int t = 0; t < 8; ++t)
            acc[t] = (v8f){0.f, 0.f, 0.f, 0.f, 0.f, 0.f, 0.f, 0.f};

#pragma unroll
        for (int ks = 0; ks < 8; ++ks) {
#pragma unroll
            for (int t = 0; t < 8; ++t) {
                // B layout: lane n = ln, element i -> k = (i%8) + 8*hi + 16*(i/8)
                Frag16 bfr;
                const unsigned short* p =
                    bbase + (t * 16 + ln) * CC + ks * 32 + 8 * hi;
                bfr.u[0] = *(const u32x4*)(p);
                bfr.u[1] = *(const u32x4*)(p + 16);
                acc[t] = __builtin_amdgcn_wmma_f32_16x16x32_bf16(
                    false, afr[ks].v, false, bfr.v, (short)0, acc[t],
                    false, false);
            }
        }

        // score = ||e||^2 - 2*(z.e); running argmin (cols ascend -> first-min kept)
#pragma unroll
        for (int t = 0; t < 8; ++t) {
            const int col = col0 + t * 16 + ln;
            const float en = enorm[col];
#pragma unroll
            for (int r = 0; r < 8; ++r) {
                float s = en - 2.0f * acc[t][r];
                if (s < bestv[r] || (s == bestv[r] && col < bestc[r])) {
                    bestv[r] = s; bestc[r] = col;
                }
            }
        }
        __syncthreads();       // all reads done before buffer is re-DMA'd
    }

    // reduce over the 16 lanes of each half-wave; D row = r + 8*hi
#pragma unroll
    for (int r = 0; r < 8; ++r) {
        float v = bestv[r];
        int   c = bestc[r];
        for (int off = 8; off >= 1; off >>= 1) {
            float ov = __shfl_xor(v, off, 16);
            int   oc = __shfl_xor(c, off, 16);
            if (ov < v || (ov == v && oc < c)) { v = ov; c = oc; }
        }
        if (ln == 0) d_idx[row0 + r + 8 * hi] = c;
    }
}

// ------------------------------------------------ 3: gather/z_q/loss/one-hot/idx
// grid = 1024 blocks (b * 32 l-chunks), 256 threads
__global__ __launch_bounds__(256)
void vq_outputs(const float* __restrict__ z,
                const float* __restrict__ maskp,
                const float* __restrict__ emb,
                const int* __restrict__ d_idx,
                float* __restrict__ out,
                int* __restrict__ e_count,
                float* __restrict__ ws_loss) {
    __shared__ int   sidx[32];
    __shared__ float sred[256];
    const int b  = blockIdx.x >> 5;
    const int l0 = (blockIdx.x & 31) * 32;
    const int t  = threadIdx.x;
    if (t < 32) sidx[t] = d_idx[b * LL + l0 + t];
    __syncthreads();

    const int lt = t & 31;
    const int ct = t >> 5;
    const float m = maskp[b * LL + l0 + lt];
    const int   e = sidx[lt];
    float acc = 0.f;

    float* zq = out + O_ZQ;
#pragma unroll 4
    for (int cc = ct; cc < CC; cc += 8) {
        float q  = emb[e * CC + cc];
        size_t o = (size_t)b * CC * LL + (size_t)cc * LL + l0 + lt;
        float zv = z[o];
        zq[o] = q;                       // straight-through forward value == z_q
        float d = (q - zv) * m;
        acc += d * d;
    }
    sred[t] = acc;
    __syncthreads();
    for (int s = 128; s > 0; s >>= 1) {
        if (t < s) sred[t] += sred[t + s];
        __syncthreads();
    }
    if (t == 0) atomicAdd(ws_loss, sred[0]);

    if (t < 32) {
        int n = b * LL + l0 + t;
        int ei = sidx[t];
        out[O_OH + (size_t)n * NE + ei] = 1.0f;
        atomicAdd(&e_count[ei], 1);
        out[O_IDX + n] = (float)ei;
    }
}

// ------------------------------------------------ 4: perplexity + loss finalize
__global__ __launch_bounds__(1024)
void vq_finalize(const int* __restrict__ e_count,
                 const float* __restrict__ ws_loss,
                 float* __restrict__ out) {
    __shared__ float sred[1024];
    const int t = threadIdx.x;
    float em = (float)e_count[t] * (1.0f / (float)NN);
    sred[t] = em * logf(em + 1e-10f);
    __syncthreads();
    for (int s = 512; s > 0; s >>= 1) {
        if (t < s) sred[t] += sred[t + s];
        __syncthreads();
    }
    if (t == 0) {
        out[O_PERP] = expf(-sred[0]);
        out[O_LOSS] = (1.0f + BETA) * ws_loss[0] /
                      (float)((size_t)BB * LL * CC);
    }
}

// ---------------------------------------------------------------- launcher
extern "C" void kernel_launch(void* const* d_in, const int* in_sizes, int n_in,
                              void* d_out, int out_size, void* d_ws, size_t ws_size,
                              hipStream_t stream) {
    const float* z    = (const float*)d_in[0];
    const float* mask = (const float*)d_in[1];
    const float* emb  = (const float*)d_in[2];
    float* out = (float*)d_out;
    char*  ws  = (char*)d_ws;

    unsigned short* zbf   = (unsigned short*)(ws + WS_ZBF);
    unsigned short* ebf   = (unsigned short*)(ws + WS_EBF);
    float*          enorm = (float*)(ws + WS_ENORM);
    int*            didx  = (int*)(ws + WS_IDX);
    int*            ecnt  = (int*)(ws + WS_CNT);
    float*          wloss = (float*)(ws + WS_LOSS);

    // zero the counters/loss accumulator and the one-hot output region
    (void)hipMemsetAsync(ws + WS_CNT, 0, (NE + 1) * sizeof(float) + 64, stream);
    (void)hipMemsetAsync(out + O_OH, 0, (size_t)NN * NE * sizeof(float), stream);

    vq_prep_embed<<<NE, 256, 0, stream>>>(emb, ebf, enorm);
    vq_transpose_z<<<dim3(LL / 32, CC / 32, BB), dim3(32, 8), 0, stream>>>(z, zbf);
    vq_score_argmin<<<NN / 128, 256, 0, stream>>>(zbf, ebf, enorm, didx);
    vq_outputs<<<BB * (LL / 32), 256, 0, stream>>>(z, mask, emb, didx, out, ecnt, wloss);
    vq_finalize<<<1, 1024, 0, stream>>>(ecnt, wloss, out);
}
